// GRU_Topology_89945205113502
// MI455X (gfx1250) — matrix operand order repaired and checked by reading, not compiled, not run on hardware
//
#include <hip/hip_runtime.h>

// ---------------- problem constants ----------------
#define MM    256   // hidden
#define DIMF  128   // feature dim
#define TTS   128   // timesteps
#define NWG   16    // B / 16

// ---------------- vector types ----------------
typedef __bf16 v16bf __attribute__((ext_vector_type(16)));
typedef __bf16 v8bf  __attribute__((ext_vector_type(8)));
typedef float  v8f   __attribute__((ext_vector_type(8)));
typedef float  v4f   __attribute__((ext_vector_type(4)));
typedef unsigned int v4u __attribute__((ext_vector_type(4)));

// ---------------- workspace layout (bytes) ----------------
#define OFF_WZR 0u
#define SZ_WZR  (32u*12u*1024u)      // 393216
#define OFF_WH1 (OFF_WZR + SZ_WZR)
#define SZ_WH1  (16u*8u*1024u)       // 131072
#define OFF_WH2 (OFF_WH1 + SZ_WH1)
#define SZ_WH2  (16u*4u*1024u)       //  65536
#define OFF_XB  (OFF_WH2 + SZ_WH2)   // X B-frags: T x NWG x 4KB = 8388608

// LDS strides (padded: +4 banks per column)
#define HSTR 264   // ushort units per column (528B)
#define FSTR 260   // float  units per column (1040B)

__device__ __forceinline__ unsigned short f2bf(float f) {
    __bf16 b = (__bf16)f;
    union { __bf16 b; unsigned short u; } cv; cv.b = b; return cv.u;
}
__device__ __forceinline__ unsigned pk2(float lo, float hi) {
    return ((unsigned)f2bf(hi) << 16) | (unsigned)f2bf(lo);
}
__device__ __forceinline__ float sigm(float x)      { return 1.f / (1.f + __expf(-x)); }
__device__ __forceinline__ float tanh_fast(float x) { return 1.f - 2.f / (__expf(2.f * x) + 1.f); }

// 16x32 bf16 B-fragment from col-major LDS: two ds_load_b128 off one base
__device__ __forceinline__ v16bf lds_bfrag(const unsigned short* p) {
    v8bf lo = *(const v8bf*)p;
    v8bf hi = *(const v8bf*)(p + 8);
    v16bf v;
#pragma unroll
    for (int i = 0; i < 8; ++i) { v[i] = lo[i]; v[i + 8] = hi[i]; }
    return v;
}

// ============ prologue 1: weights -> bf16 A-fragment order ============
__global__ __launch_bounds__(256) void pack_weights(
    const float* __restrict__ Wz, const float* __restrict__ Wr,
    const float* __restrict__ Wh1, const float* __restrict__ Wh2,
    unsigned short* __restrict__ wzr, unsigned short* __restrict__ wh1p,
    unsigned short* __restrict__ wh2p)
{
    const int NZR = 32 * 12 * 512, NH1 = 16 * 8 * 512, NH2 = 16 * 4 * 512;
    int o = blockIdx.x * 256 + threadIdx.x;
    if (o < NZR) {
        int tile = o >> 9, r = o & 511, lane = r >> 4, i = r & 15;
        int mt = tile / 12, kt = tile % 12;
        int row = mt * 16 + (lane & 15);
        int b0 = (lane < 16) ? 0 : 8;
        int k = kt * 32 + 2 * b0 + i;
        float v = (row < 256) ? Wz[row * 384 + k] : Wr[(row - 256) * 384 + k];
        wzr[o] = f2bf(v);
    } else if (o < NZR + NH1) {
        int o2 = o - NZR;
        int tile = o2 >> 9, r = o2 & 511, lane = r >> 4, i = r & 15;
        int mt = tile >> 3, kt = tile & 7;
        int row = mt * 16 + (lane & 15);
        int b0 = (lane < 16) ? 0 : 8;
        int k = kt * 32 + 2 * b0 + i;
        wh1p[o2] = f2bf(Wh1[row * 256 + k]);
    } else if (o < NZR + NH1 + NH2) {
        int o3 = o - NZR - NH1;
        int tile = o3 >> 9, r = o3 & 511, lane = r >> 4, i = r & 15;
        int mt = tile >> 2, kt = tile & 3;
        int row = mt * 16 + (lane & 15);
        int b0 = (lane < 16) ? 0 : 8;
        int k = kt * 32 + 2 * b0 + i;
        wh2p[o3] = f2bf(Wh2[row * 128 + k]);
    }
}

// ============ prologue 2: gather topo[x], emit bf16 B-fragments ============
// Layout: [t][wg][kt(4)][lane(32)][16] bf16 => 4KB per (t,wg)
__global__ __launch_bounds__(256) void gather_x(
    const int* __restrict__ x, const float* __restrict__ topo,
    unsigned short* __restrict__ xb)
{
    unsigned o = blockIdx.x * 256u + threadIdx.x;   // < 4194304
    int t    = o >> 15;
    int rem  = o & 32767;
    int wg   = rem >> 11;
    int e    = rem & 2047;
    int kt   = e >> 9;
    int r    = e & 511;
    int lane = r >> 4, i = r & 15;
    int col  = wg * 16 + (lane & 15);
    int k    = kt * 32 + ((lane < 16) ? 0 : 16) + i;
    int idx  = x[col];
    xb[o] = f2bf(topo[((size_t)idx * TTS + t) * DIMF + k]);
}

// ============ main recurrent kernel: 16 WGs x 512 threads ============
__global__ __launch_bounds__(512) void gru_main(
    const float* __restrict__ bz, const float* __restrict__ br, const float* __restrict__ bh,
    const unsigned short* __restrict__ wzr, const unsigned short* __restrict__ wh1,
    const unsigned short* __restrict__ wh2, const unsigned short* __restrict__ xb,
    float* __restrict__ out)
{
    __shared__ __attribute__((aligned(16))) unsigned short hb[16 * HSTR]; // h bf16 (B-frag)
    __shared__ __attribute__((aligned(16))) unsigned short rb[16 * HSTR]; // r*h bf16
    __shared__ __attribute__((aligned(16))) float hf[16 * FSTR];          // h f32
    __shared__ __attribute__((aligned(16))) float zf[16 * FSTR];          // z f32
    __shared__ __attribute__((aligned(16))) float rf[16 * FSTR];          // r f32

    const int tid  = threadIdx.x;
    const int wave = tid >> 5;       // 0..15
    const int lane = tid & 31;
    const int col  = lane & 15;
    const int hlf  = lane >> 4;
    const int wg   = blockIdx.x;
    const int rowoff = hlf * 8;
    const int rowb   = wave * 16 + rowoff;

    // ---- t-invariant base pointers (all address math hoisted) ----
    const int mt0 = wave * 2;
    const v16bf* azr = (const v16bf*)wzr + mt0 * 384 + lane;   // a1 at +384
    const v16bf* ah1 = (const v16bf*)wh1 + wave * 256 + lane;
    const v16bf* ah2 = (const v16bf*)wh2 + wave * 128 + lane;
    const v16bf* xt  = (const v16bf*)xb  + wg * 128 + lane;    // += 2048 per t

    const unsigned short* hbp = hb + col * HSTR + hlf * 16;    // + kt*32 imm
    const unsigned short* rbp = rb + col * HSTR + hlf * 16;

    float* zrdst0 = ((mt0     < 16) ? zf : rf) + col * FSTR + ((mt0    ) & 15) * 16 + rowoff;
    float* zrdst1 = ((mt0 + 1 < 16) ? zf : rf) + col * FSTR + ((mt0 + 1) & 15) * 16 + rowoff;

    const int c3 = tid & 15, q3 = tid >> 4;                    // phase-3 mapping
    const float* rp3 = rf + c3 * FSTR + q3 * 8;
    const float* hp3 = hf + c3 * FSTR + q3 * 8;
    unsigned short* rb3 = rb + c3 * HSTR + q3 * 8;

    const float* zp5 = zf + col * FSTR + rowb;
    float*       hf5 = hf + col * FSTR + rowb;                 // read then overwrite
    unsigned short* hb6 = hb + col * HSTR + rowb;
    float* outp = out + ((size_t)(wg * 16 + col) * TTS) * MM + rowb;  // += MM per t

    // ---- hoisted biases ----
    float bZR[2][8];
#pragma unroll
    for (int j = 0; j < 2; ++j) {
        int mt = mt0 + j;
        const float* p = ((mt < 16) ? bz : br) + (mt & 15) * 16 + rowoff;
        v4f a = *(const v4f*)p, b2 = *(const v4f*)(p + 4);
#pragma unroll
        for (int i = 0; i < 4; ++i) { bZR[j][i] = a[i]; bZR[j][i + 4] = b2[i]; }
    }
    float bH[8];
    {
        const float* p = bh + wave * 16 + rowoff;
        v4f a = *(const v4f*)p, b2 = *(const v4f*)(p + 4);
#pragma unroll
        for (int i = 0; i < 4; ++i) { bH[i] = a[i]; bH[i + 4] = b2[i]; }
    }

    for (int t = 0; t < TTS; ++t) {
        if (t + 1 < TTS) {
#pragma unroll
            for (int j = 0; j < 4; ++j) __builtin_prefetch(xt + 2048 + j * 32, 0, 1);
        }

        if (t > 0) {
            // ---- phase 1: ZR = [Wz;Wr] @ [h; x] ----
            v8f acc0 = {0.f,0.f,0.f,0.f,0.f,0.f,0.f,0.f};
            v8f acc1 = {0.f,0.f,0.f,0.f,0.f,0.f,0.f,0.f};
#pragma unroll 4
            for (int kt = 0; kt < 8; ++kt) {
                v16bf bfr = lds_bfrag(hbp + kt * 32);
                v16bf a0  = azr[kt * 32];
                v16bf a1  = azr[384 + kt * 32];
                acc0 = __builtin_amdgcn_wmma_f32_16x16x32_bf16(false, a0, false, bfr, (short)0, acc0, false, false);
                acc1 = __builtin_amdgcn_wmma_f32_16x16x32_bf16(false, a1, false, bfr, (short)0, acc1, false, false);
            }
#pragma unroll 2
            for (int j = 0; j < 4; ++j) {
                v16bf bfr = xt[j * 32];
                v16bf a0  = azr[(8 + j) * 32];
                v16bf a1  = azr[384 + (8 + j) * 32];
                acc0 = __builtin_amdgcn_wmma_f32_16x16x32_bf16(false, a0, false, bfr, (short)0, acc0, false, false);
                acc1 = __builtin_amdgcn_wmma_f32_16x16x32_bf16(false, a1, false, bfr, (short)0, acc1, false, false);
            }
            // ---- phase 2: sigmoid -> z/r (f32 LDS) ----
#pragma unroll
            for (int j = 0; j < 2; ++j) {
                v8f& a = (j == 0) ? acc0 : acc1;
                float* dst = (j == 0) ? zrdst0 : zrdst1;
                v4f sA, sB;
#pragma unroll
                for (int i = 0; i < 4; ++i) {
                    sA[i] = sigm(a[i]     + bZR[j][i]);
                    sB[i] = sigm(a[i + 4] + bZR[j][i + 4]);
                }
                *(v4f*)dst = sA;
                *(v4f*)(dst + 4) = sB;
            }
            __syncthreads();
            // ---- phase 3: rh = r * h -> bf16 B-frag layout ----
            {
                v4f r0 = *(const v4f*)rp3, r1 = *(const v4f*)(rp3 + 4);
                v4f h0 = *(const v4f*)hp3, h1 = *(const v4f*)(hp3 + 4);
                v4u pkv;
                pkv[0] = pk2(r0[0] * h0[0], r0[1] * h0[1]);
                pkv[1] = pk2(r0[2] * h0[2], r0[3] * h0[3]);
                pkv[2] = pk2(r1[0] * h1[0], r1[1] * h1[1]);
                pkv[3] = pk2(r1[2] * h1[2], r1[3] * h1[3]);
                *(v4u*)rb3 = pkv;
            }
            __syncthreads();
        }

        // ---- phase 4: h~ = Wh1 @ rh + Wh2 @ x + bh ----
        v8f acc = {0.f,0.f,0.f,0.f,0.f,0.f,0.f,0.f};
        if (t > 0) {
#pragma unroll 4
            for (int kt = 0; kt < 8; ++kt) {
                v16bf bfr = lds_bfrag(rbp + kt * 32);
                v16bf a   = ah1[kt * 32];
                acc = __builtin_amdgcn_wmma_f32_16x16x32_bf16(false, a, false, bfr, (short)0, acc, false, false);
            }
        }
#pragma unroll 2
        for (int j = 0; j < 4; ++j) {
            v16bf bfr = xt[j * 32];
            v16bf a   = ah2[j * 32];
            acc = __builtin_amdgcn_wmma_f32_16x16x32_bf16(false, a, false, bfr, (short)0, acc, false, false);
        }

        // ---- phase 5: gate combine ----
        float ht[8];
        if (t > 0) {
            v4f z0 = *(const v4f*)zp5, z1 = *(const v4f*)(zp5 + 4);
            v4f h0 = *(const v4f*)hf5, h1 = *(const v4f*)(hf5 + 4);
#pragma unroll
            for (int i = 0; i < 4; ++i) {
                float hti = tanh_fast(acc[i] + bH[i]);
                ht[i] = h0[i] + z0[i] * (hti - h0[i]);
                float htj = tanh_fast(acc[i + 4] + bH[i + 4]);
                ht[i + 4] = h1[i] + z1[i] * (htj - h1[i]);
            }
        } else {
#pragma unroll
            for (int i = 0; i < 8; ++i) ht[i] = tanh_fast(acc[i] + bH[i]);
        }

        // ---- phase 6: write out / hf / hb ----
        v4f o0, o1;
#pragma unroll
        for (int i = 0; i < 4; ++i) { o0[i] = ht[i]; o1[i] = ht[i + 4]; }
        *(v4f*)outp = o0;
        *(v4f*)(outp + 4) = o1;
        *(v4f*)hf5 = o0;
        *(v4f*)(hf5 + 4) = o1;
        v4u hpk;
        hpk[0] = pk2(ht[0], ht[1]); hpk[1] = pk2(ht[2], ht[3]);
        hpk[2] = pk2(ht[4], ht[5]); hpk[3] = pk2(ht[6], ht[7]);
        *(v4u*)hb6 = hpk;

        xt   += 2048;   // next t slab (16 wg * 128 v16bf)
        outp += MM;
        __syncthreads();
    }
}

// ============================ launch ============================
extern "C" void kernel_launch(void* const* d_in, const int* in_sizes, int n_in,
                              void* d_out, int out_size, void* d_ws, size_t ws_size,
                              hipStream_t stream) {
    const int*   x    = (const int*)d_in[0];
    const float* topo = (const float*)d_in[1];
    const float* Wz   = (const float*)d_in[2];
    const float* Wr   = (const float*)d_in[3];
    const float* Wh1  = (const float*)d_in[4];
    const float* Wh2  = (const float*)d_in[5];
    const float* bz   = (const float*)d_in[6];
    const float* br   = (const float*)d_in[7];
    const float* bh   = (const float*)d_in[8];
    float* out = (float*)d_out;

    char* ws = (char*)d_ws;
    unsigned short* wzr  = (unsigned short*)(ws + OFF_WZR);
    unsigned short* wh1p = (unsigned short*)(ws + OFF_WH1);
    unsigned short* wh2p = (unsigned short*)(ws + OFF_WH2);
    unsigned short* xb   = (unsigned short*)(ws + OFF_XB);

    pack_weights<<<1152, 256, 0, stream>>>(Wz, Wr, Wh1, Wh2, wzr, wh1p, wh2p);
    gather_x<<<16384, 256, 0, stream>>>(x, topo, xb);
    gru_main<<<NWG, 512, 0, stream>>>(bz, br, bh, wzr, wh1p, wh2p, xb, out);
}
